// GetModel_15238543966696
// MI455X (gfx1250) — matrix operand order repaired and checked by reading, not compiled, not run on hardware
//
#include <hip/hip_runtime.h>
#include <hip/hip_fp16.h>

// PointNet++ MSG part-seg forward for MI455X (gfx1250).
// Matmul layers: v_wmma_f32_16x16x32_f16, 2 n-tiles per wave, weights staged
// into LDS via global_load_async_to_lds_b128 (ASYNCcnt) shared by all 8 waves.

typedef __attribute__((ext_vector_type(8)))  _Float16 h8;
typedef __attribute__((ext_vector_type(16))) _Float16 v16h;
typedef __attribute__((ext_vector_type(8)))  float    v8f;

#define BATCH 16
#define NPTS  4096

static inline size_t align256(size_t x){ return (x + 255) & ~(size_t)255; }
static inline int ceil32(int x){ return (x + 31) & ~31; }

// ---------------------------------------------------------------- transpose
__global__ void k_transpose_xyz(const float* __restrict__ xyz, float* __restrict__ out,
                                int B, int N){
  int i = blockIdx.x * blockDim.x + threadIdx.x;
  if (i >= B * N) return;
  int b = i / N, n = i % N;
  out[(size_t)i * 3 + 0] = xyz[((size_t)b * 3 + 0) * N + n];
  out[(size_t)i * 3 + 1] = xyz[((size_t)b * 3 + 1) * N + n];
  out[(size_t)i * 3 + 2] = xyz[((size_t)b * 3 + 2) * N + n];
}

// ---------------------------------------------------------------- FPS
__global__ void k_fps(const float* __restrict__ xyz, float* __restrict__ dist,
                      int* __restrict__ fidx, float* __restrict__ newxyz,
                      int N, int npoint){
  int b = blockIdx.x;
  const float* px = xyz + (size_t)b * N * 3;
  float* dw = dist + (size_t)b * N;
  __shared__ float sval[256];
  __shared__ int   sidx[256];
  __shared__ int   s_far;
  int tid = threadIdx.x;

  float bv = -1e30f; int bi = 0;
  for (int n = tid; n < N; n += blockDim.x){
    dw[n] = 1e10f;
    float x = px[(size_t)n * 3];
    if (x > bv){ bv = x; bi = n; }
  }
  sval[tid] = bv; sidx[tid] = bi; __syncthreads();
  for (int s = blockDim.x / 2; s > 0; s >>= 1){
    if (tid < s && sval[tid + s] > sval[tid]){ sval[tid] = sval[tid + s]; sidx[tid] = sidx[tid + s]; }
    __syncthreads();
  }
  if (tid == 0) s_far = sidx[0];
  __syncthreads();

  for (int it = 0; it < npoint; ++it){
    int far = s_far;
    if (tid == 0) fidx[b * npoint + it] = far;
    float cx = px[(size_t)far * 3], cy = px[(size_t)far * 3 + 1], cz = px[(size_t)far * 3 + 2];
    float mv = -1.0f; int mi = 0;
    for (int n = tid; n < N; n += blockDim.x){
      float dx = px[(size_t)n * 3] - cx, dy = px[(size_t)n * 3 + 1] - cy, dz = px[(size_t)n * 3 + 2] - cz;
      float d = dx * dx + dy * dy + dz * dz;
      float old = dw[n];
      d = d < old ? d : old;
      dw[n] = d;
      if (d > mv){ mv = d; mi = n; }
    }
    sval[tid] = mv; sidx[tid] = mi; __syncthreads();
    for (int s = blockDim.x / 2; s > 0; s >>= 1){
      if (tid < s && sval[tid + s] > sval[tid]){ sval[tid] = sval[tid + s]; sidx[tid] = sidx[tid + s]; }
      __syncthreads();
    }
    if (tid == 0) s_far = sidx[0];
    __syncthreads();
  }
  for (int s = tid; s < npoint; s += blockDim.x){
    int id = fidx[b * npoint + s];
    newxyz[((size_t)b * npoint + s) * 3 + 0] = px[(size_t)id * 3 + 0];
    newxyz[((size_t)b * npoint + s) * 3 + 1] = px[(size_t)id * 3 + 1];
    newxyz[((size_t)b * npoint + s) * 3 + 2] = px[(size_t)id * 3 + 2];
  }
}

// ---------------------------------------------------------------- ball query
__global__ void k_ballquery(const float* __restrict__ sx, const float* __restrict__ cx,
                            int* __restrict__ gidx, int B, int Nsrc, int np, int K, float r2){
  int i = blockIdx.x * blockDim.x + threadIdx.x;
  if (i >= B * np) return;
  int b = i / np;
  const float* p = sx + (size_t)b * Nsrc * 3;
  float qx = cx[(size_t)i * 3], qy = cx[(size_t)i * 3 + 1], qz = cx[(size_t)i * 3 + 2];
  int* out = gidx + (size_t)i * K;
  int cnt = 0;
  for (int n = 0; n < Nsrc && cnt < K; ++n){
    float dx = p[(size_t)n * 3] - qx, dy = p[(size_t)n * 3 + 1] - qy, dz = p[(size_t)n * 3 + 2] - qz;
    if (dx * dx + dy * dy + dz * dz <= r2) out[cnt++] = n;
  }
  int f = cnt > 0 ? out[0] : 0;
  for (int k = cnt; k < K; ++k) out[k] = f;
}

// ---------------------------------------------------------------- group build (one batch)
__global__ void k_build_group(const float* __restrict__ sx, const float* __restrict__ cx,
                              const int* __restrict__ gidx, const _Float16* __restrict__ feat,
                              _Float16* __restrict__ out,
                              int b, int Nsrc, int np, int K, int Cfeat, int Kpad){
  int r = blockIdx.x * blockDim.x + threadIdx.x;     // row within this batch: np*K
  if (r >= np * K) return;
  int s = r / K;
  int gi = gidx[((size_t)b * np + s) * K + (r % K)];
  const float* p = sx + ((size_t)b * Nsrc + gi) * 3;
  const float* c = cx + ((size_t)b * np + s) * 3;
  _Float16* o = out + (size_t)r * Kpad;
  int cpos = 0;
  if (feat){
    const _Float16* f = feat + ((size_t)b * Nsrc + gi) * Cfeat;
    for (; cpos < Cfeat; ++cpos) o[cpos] = f[cpos];
  } else {
    o[0] = (_Float16)p[0]; o[1] = (_Float16)p[1]; o[2] = (_Float16)p[2]; cpos = 3;
  }
  o[cpos++] = (_Float16)(p[0] - c[0]);
  o[cpos++] = (_Float16)(p[1] - c[1]);
  o[cpos++] = (_Float16)(p[2] - c[2]);
  for (; cpos < Kpad; ++cpos) o[cpos] = (_Float16)0.f;
}

// ---------------------------------------------------------------- weight/bias prep (f32 -> padded f16)
__global__ void k_prep_w(const float* __restrict__ W, _Float16* __restrict__ o,
                         int Nout, int Kin, int Npad, int Kpad){
  int i = blockIdx.x * blockDim.x + threadIdx.x;
  if (i >= Npad * Kpad) return;
  int n = i / Kpad, k = i % Kpad;
  o[i] = (n < Nout && k < Kin) ? (_Float16)W[(size_t)n * Kin + k] : (_Float16)0.f;
}
__global__ void k_prep_b(const float* __restrict__ Bv, float* __restrict__ o, int Nout, int Npad){
  int i = blockIdx.x * blockDim.x + threadIdx.x;
  if (i >= Npad) return;
  o[i] = i < Nout ? Bv[i] : 0.f;
}

// ---------------------------------------------------------------- WMMA GEMM
// D = [relu](A[M,Kp] * W[Np,Kp]^T + b). One wave -> 16x32 output (2 n-tiles,
// A fragment reused). The 32-column weight slab (shared by all 8 waves) is
// staged to LDS with global_load_async_to_lds_b128 and read back as fragments.
__global__ void __launch_bounds__(256)
k_gemm(const _Float16* __restrict__ A, const _Float16* __restrict__ W,
       const float* __restrict__ bias, _Float16* __restrict__ outH, float* __restrict__ outF,
       int M, int Kp, int Np, int relu){
  extern __shared__ _Float16 smem[];          // 32 cols x (Kp+16) halves
  const int ldsStride = Kp + 16;              // +32B pad: de-conflict LDS banks
  const int n0 = blockIdx.y * 32;

  // --- async bulk copy: W[n0..n0+32) x Kp  ->  LDS (per-column, padded)
  {
    const uint32_t lds0 = (uint32_t)(uintptr_t)smem;
    const _Float16* src = W + (size_t)n0 * Kp;        // slab is contiguous in W
    int chunksPerCol = Kp >> 3;                       // 16B chunks per column
    int total = 32 * chunksPerCol;
    for (int c = threadIdx.x; c < total; c += 256){
      int col = c / chunksPerCol;
      int rem = c - col * chunksPerCol;
      uint32_t goff = (uint32_t)((col * Kp + rem * 8) * 2);
      uint32_t ldsa = lds0 + (uint32_t)((col * ldsStride + rem * 8) * 2);
      asm volatile("global_load_async_to_lds_b128 %0, %1, %2"
                   :: "v"(ldsa), "v"(goff), "s"(src) : "memory");
    }
    asm volatile("s_wait_asynccnt 0x0" ::: "memory");
  }
  __syncthreads();

  int wave = threadIdx.x >> 5;
  int lane = threadIdx.x & 31;
  int m0 = (blockIdx.x * 8 + wave) * 16;
  if (m0 >= M) return;                        // uniform per wave
  int l  = lane & 15;
  int hi = lane >> 4;

  const _Float16* arow = A + (size_t)(m0 + l) * Kp + hi * 8;        // A rows
  const _Float16* w0 = smem + (size_t)l * ldsStride + hi * 8;      // cols n0+l
  const _Float16* w1 = smem + (size_t)(16 + l) * ldsStride + hi * 8; // cols n0+16+l

  v8f acc0 = {}, acc1 = {};
  for (int k0 = 0; k0 < Kp; k0 += 32){
    h8 x = *(const h8*)(arow + k0);           // K: hi? 8..15 : 0..7
    h8 y = *(const h8*)(arow + k0 + 16);      // K: hi? 24..31 : 16..23
    v16h af;
#pragma unroll
    for (int i = 0; i < 8; ++i){ af[i] = x[i]; af[8 + i] = y[i]; }
    h8 b0a = *(const h8*)(w0 + k0);
    h8 b0b = *(const h8*)(w0 + k0 + 16);
    h8 b1a = *(const h8*)(w1 + k0);
    h8 b1b = *(const h8*)(w1 + k0 + 16);
    v16h bf0, bf1;
#pragma unroll
    for (int i = 0; i < 8; ++i){ bf0[i] = b0a[i]; bf0[8 + i] = b0b[i];
                                 bf1[i] = b1a[i]; bf1[8 + i] = b1b[i]; }
    acc0 = __builtin_amdgcn_wmma_f32_16x16x32_f16(false, af, false, bf0,
                                                  (short)0, acc0, false, false);
    acc1 = __builtin_amdgcn_wmma_f32_16x16x32_f16(false, af, false, bf1,
                                                  (short)0, acc1, false, false);
  }
  int na = n0 + l, nb = n0 + 16 + l;
  float bva = bias[na], bvb = bias[nb];
#pragma unroll
  for (int r = 0; r < 8; ++r){
    float va = acc0[r] + bva;
    float vb = acc1[r] + bvb;
    if (relu){ va = va > 0.f ? va : 0.f; vb = vb > 0.f ? vb : 0.f; }
    size_t row = (size_t)(m0 + r + hi * 8) * (size_t)Np;
    if (outH){ outH[row + na] = (_Float16)va; outH[row + nb] = (_Float16)vb; }
    else     { outF[row + na] = va;           outF[row + nb] = vb; }
  }
}

// ---------------------------------------------------------------- max over group dim
__global__ void k_maxpool(const _Float16* __restrict__ in, _Float16* __restrict__ out,
                          int G, int K, int C, int Win, int Wout, int cofs){
  int i = blockIdx.x * blockDim.x + threadIdx.x;   // over G*C
  if (i >= G * C) return;
  int g = i / C, c = i % C;
  const _Float16* p = in + (size_t)g * K * Win + c;
  float m = -1e30f;
  for (int k = 0; k < K; ++k){ float v = (float)p[(size_t)k * Win]; m = v > m ? v : m; }
  out[(size_t)g * Wout + cofs + c] = (_Float16)m;
}

// ---------------------------------------------------------------- sa3 / fp3 input builders
__global__ void k_build_sa3(const float* __restrict__ l2xyz, const _Float16* __restrict__ l2p,
                            _Float16* __restrict__ out, int rows, int Kpad){
  int r = blockIdx.x * blockDim.x + threadIdx.x;
  if (r >= rows) return;
  _Float16* o = out + (size_t)r * Kpad;
  const float* x = l2xyz + (size_t)r * 3;
  o[0] = (_Float16)x[0]; o[1] = (_Float16)x[1]; o[2] = (_Float16)x[2];
  const _Float16* f = l2p + (size_t)r * 512;
  for (int c = 0; c < 512; ++c) o[3 + c] = f[c];
  for (int c = 515; c < Kpad; ++c) o[c] = (_Float16)0.f;
}

__global__ void k_build_fp3(const _Float16* __restrict__ l2p, const _Float16* __restrict__ l3,
                            _Float16* __restrict__ out, int B, int S, int Kpad){
  int r = blockIdx.x * blockDim.x + threadIdx.x;
  if (r >= B * S) return;
  int b = r / S;
  _Float16* o = out + (size_t)r * Kpad;
  const _Float16* f = l2p + (size_t)r * 512;
  for (int c = 0; c < 512; ++c) o[c] = f[c];
  const _Float16* g = l3 + (size_t)b * 1024;
  for (int c = 0; c < 1024; ++c) o[512 + c] = g[c];
}

// ---------------------------------------------------------------- 3-NN interpolation + concat build
__global__ void k_interp_build(const float* __restrict__ xd, const float* __restrict__ xs,
                               const _Float16* __restrict__ f1, int C1,
                               const _Float16* __restrict__ fs, int Cs,
                               _Float16* __restrict__ out, int B, int Nd, int Ns, int Kpad){
  int i = blockIdx.x * blockDim.x + threadIdx.x;
  if (i >= B * Nd) return;
  int b = i / Nd;
  const float* q = xd + (size_t)i * 3;
  const float* p = xs + (size_t)b * Ns * 3;
  float d0 = 1e30f, d1 = 1e30f, d2 = 1e30f; int i0 = 0, i1 = 0, i2 = 0;
  for (int n = 0; n < Ns; ++n){
    float dx = p[(size_t)n * 3] - q[0], dy = p[(size_t)n * 3 + 1] - q[1], dz = p[(size_t)n * 3 + 2] - q[2];
    float d = dx * dx + dy * dy + dz * dz;
    if (d < d0){ d2 = d1; i2 = i1; d1 = d0; i1 = i0; d0 = d; i0 = n; }
    else if (d < d1){ d2 = d1; i2 = i1; d1 = d; i1 = n; }
    else if (d < d2){ d2 = d; i2 = n; }
  }
  float w0 = 1.f / (d0 + 1e-8f), w1 = 1.f / (d1 + 1e-8f), w2 = 1.f / (d2 + 1e-8f);
  float wsum = w0 + w1 + w2; w0 /= wsum; w1 /= wsum; w2 /= wsum;
  _Float16* o = out + (size_t)i * Kpad;
  const _Float16* a = f1 + (size_t)i * C1;
  int c = 0;
  for (; c < C1; ++c) o[c] = a[c];
  const _Float16* s0 = fs + ((size_t)b * Ns + i0) * Cs;
  const _Float16* s1 = fs + ((size_t)b * Ns + i1) * Cs;
  const _Float16* s2 = fs + ((size_t)b * Ns + i2) * Cs;
  for (int cc = 0; cc < Cs; ++cc)
    o[c + cc] = (_Float16)(w0 * (float)s0[cc] + w1 * (float)s1[cc] + w2 * (float)s2[cc]);
  for (c += Cs; c < Kpad; ++c) o[c] = (_Float16)0.f;
}

// ---------------------------------------------------------------- p1 = [cls_onehot(16), xyz(3), xyz(3)]
__global__ void k_build_p1(const float* __restrict__ cls, const float* __restrict__ l0xyz,
                           _Float16* __restrict__ out, int B, int N){
  int i = blockIdx.x * blockDim.x + threadIdx.x;
  if (i >= B * N) return;
  int b = i / N;
  _Float16* o = out + (size_t)i * 22;
  for (int c = 0; c < 16; ++c) o[c] = (_Float16)cls[b * 16 + c];
  const float* x = l0xyz + (size_t)i * 3;
  o[16] = (_Float16)x[0]; o[17] = (_Float16)x[1]; o[18] = (_Float16)x[2];
  o[19] = (_Float16)x[0]; o[20] = (_Float16)x[1]; o[21] = (_Float16)x[2];
}

// ---------------------------------------------------------------- log-softmax, f16->f32 copy
__global__ void k_logsoftmax(const float* __restrict__ logits, float* __restrict__ out,
                             int rows, int C, int stride){
  int i = blockIdx.x * blockDim.x + threadIdx.x;
  if (i >= rows) return;
  const float* p = logits + (size_t)i * stride;
  float m = -1e30f;
  for (int c = 0; c < C; ++c){ float v = p[c]; m = v > m ? v : m; }
  float s = 0.f;
  for (int c = 0; c < C; ++c) s += expf(p[c] - m);
  float ls = logf(s) + m;
  for (int c = 0; c < C; ++c) out[(size_t)i * C + c] = p[c] - ls;
}
__global__ void k_h2f(const _Float16* __restrict__ in, float* __restrict__ out, int n){
  int i = blockIdx.x * blockDim.x + threadIdx.x;
  if (i < n) out[i] = (float)in[i];
}

// =================================================================== host
struct LCfg { int wi, bi, Nout, Kin; };

extern "C" void kernel_launch(void* const* d_in, const int* in_sizes, int n_in,
                              void* d_out, int out_size, void* d_ws, size_t ws_size,
                              hipStream_t stream){
  (void)in_sizes; (void)n_in; (void)out_size; (void)ws_size;
  const int B = BATCH, N = NPTS;
  const float* xyz_in = (const float*)d_in[0];
  const float* cls    = (const float*)d_in[1];

  // 26 GEMM layers; d_in leaf order = jax pytree (dict keys sorted alphabetically)
  static const LCfg cfg[26] = {
    {2,3,128,128},  {4,5,50,128},                           // conv1, conv2
    {6,7,128,150},  {8,9,128,128},                          // fp1
    {10,11,256,576},{12,13,128,256},                        // fp2
    {14,15,256,1536},{16,17,256,256},                       // fp3
    {18,19,32,6},  {20,21,32,32},  {22,23,64,32},           // sa1 b0
    {24,25,64,6},  {26,27,64,64},  {28,29,128,64},          // sa1 b1
    {30,31,64,6},  {32,33,96,64},  {34,35,128,96},          // sa1 b2
    {36,37,128,323},{38,39,128,128},{40,41,256,128},        // sa2 b0
    {42,43,128,323},{44,45,196,128},{46,47,256,196},        // sa2 b1
    {48,49,256,515},{50,51,512,256},{52,53,1024,512},       // sa3
  };

  uint8_t* ws = (uint8_t*)d_ws;
  size_t off = 0;
  auto alloc = [&](size_t bytes){ size_t o = off; off = align256(off + bytes); return o; };

  size_t wofs[26], bofs[26];
  for (int i = 0; i < 26; ++i){
    int Np = ceil32(cfg[i].Nout), Kp = ceil32(cfg[i].Kin);
    wofs[i] = alloc((size_t)Np * Kp * 2);
    bofs[i] = alloc((size_t)Np * 4);
  }
  size_t O_L0XYZ = alloc((size_t)B * N * 3 * 4);
  size_t O_DIST  = alloc((size_t)B * N * 4);
  size_t O_FIDX1 = alloc((size_t)B * 512 * 4);
  size_t O_L1XYZ = alloc((size_t)B * 512 * 3 * 4);
  size_t O_FIDX2 = alloc((size_t)B * 128 * 4);
  size_t O_L2XYZ = alloc((size_t)B * 128 * 3 * 4);
  size_t O_GIDX  = alloc((size_t)B * 512 * 128 * 4);
  size_t O_L1PTS = alloc((size_t)B * 512 * 320 * 2);
  size_t O_L2PTS = alloc((size_t)B * 128 * 512 * 2);
  size_t O_L3F16 = alloc((size_t)B * 1024 * 2);
  size_t O_L2P   = alloc((size_t)B * 128 * 256 * 2);
  size_t O_L1P   = alloc((size_t)B * 512 * 128 * 2);
  size_t O_P1    = alloc((size_t)B * N * 22 * 2);
  size_t O_LOG   = alloc((size_t)B * N * 64 * 4);
  size_t O_IN    = alloc((size_t)B * N * 160 * 2);     // shared GEMM input buffer (max use)
  size_t O_ACTA  = alloc((size_t)65536 * 128 * 2);
  size_t O_ACTB  = alloc((size_t)65536 * 128 * 2);

  float*    l0xyz = (float*)(ws + O_L0XYZ);
  float*    distb = (float*)(ws + O_DIST);
  int*      fidx1 = (int*)(ws + O_FIDX1);
  float*    l1xyz = (float*)(ws + O_L1XYZ);
  int*      fidx2 = (int*)(ws + O_FIDX2);
  float*    l2xyz = (float*)(ws + O_L2XYZ);
  int*      gidx  = (int*)(ws + O_GIDX);
  _Float16* l1pts = (_Float16*)(ws + O_L1PTS);
  _Float16* l2pts = (_Float16*)(ws + O_L2PTS);
  _Float16* l3f16 = (_Float16*)(ws + O_L3F16);
  _Float16* l2p   = (_Float16*)(ws + O_L2P);
  _Float16* l1p   = (_Float16*)(ws + O_L1P);
  _Float16* p1b   = (_Float16*)(ws + O_P1);
  float*    logit = (float*)(ws + O_LOG);
  _Float16* INh   = (_Float16*)(ws + O_IN);
  _Float16* ACTA  = (_Float16*)(ws + O_ACTA);
  _Float16* ACTB  = (_Float16*)(ws + O_ACTB);

  auto gemm = [&](const _Float16* Ain, int li, int M, void* outp, bool relu, bool f16o){
    int Kp = ceil32(cfg[li].Kin), Np = ceil32(cfg[li].Nout);
    dim3 grid((M / 16 + 7) / 8, Np / 32);
    size_t sh = (size_t)32 * (Kp + 16) * 2;
    k_gemm<<<grid, 256, sh, stream>>>(Ain, (const _Float16*)(ws + wofs[li]),
                                      (const float*)(ws + bofs[li]),
                                      f16o ? (_Float16*)outp : nullptr,
                                      f16o ? nullptr : (float*)outp,
                                      M, Kp, Np, relu ? 1 : 0);
  };

  // ---- weight prep (deterministic; re-run every call)
  for (int i = 0; i < 26; ++i){
    int Np = ceil32(cfg[i].Nout), Kp = ceil32(cfg[i].Kin), nEl = Np * Kp;
    k_prep_w<<<(nEl + 255) / 256, 256, 0, stream>>>((const float*)d_in[cfg[i].wi],
        (_Float16*)(ws + wofs[i]), cfg[i].Nout, cfg[i].Kin, Np, Kp);
    k_prep_b<<<(Np + 255) / 256, 256, 0, stream>>>((const float*)d_in[cfg[i].bi],
        (float*)(ws + bofs[i]), cfg[i].Nout, Np);
  }

  // ---- l0
  k_transpose_xyz<<<(B * N + 255) / 256, 256, 0, stream>>>(xyz_in, l0xyz, B, N);
  k_fps<<<B, 256, 0, stream>>>(l0xyz, distb, fidx1, l1xyz, N, 512);

  // ---- SA1 (MSG, points = xyz): Cin 6 -> pad 32
  {
    const float radii[3] = {0.1f, 0.2f, 0.4f};
    const int   ks[3]    = {32, 64, 128};
    const int   base[3]  = {8, 11, 14};
    int cofs = 0;
    for (int br = 0; br < 3; ++br){
      int K = ks[br]; float r2 = radii[br] * radii[br];
      k_ballquery<<<(B * 512 + 127) / 128, 128, 0, stream>>>(l0xyz, l1xyz, gidx, B, N, 512, K, r2);
      int C = cfg[base[br] + 2].Nout, Win = ceil32(C);
      for (int b = 0; b < B; ++b){
        int rows = 512 * K;
        k_build_group<<<(rows + 255) / 256, 256, 0, stream>>>(l0xyz, l1xyz, gidx, nullptr,
                                                              INh, b, N, 512, K, 0, 32);
        gemm(INh,  base[br] + 0, rows, ACTA, true, true);
        gemm(ACTA, base[br] + 1, rows, ACTB, true, true);
        gemm(ACTB, base[br] + 2, rows, ACTA, true, true);
        int tot = 512 * C;
        k_maxpool<<<(tot + 255) / 256, 256, 0, stream>>>(ACTA, l1pts + (size_t)b * 512 * 320,
                                                         512, K, C, Win, 320, cofs);
      }
      cofs += C;
    }
  }

  // ---- SA2 (points = l1_points[320] + gxyz -> 323 pad 352)
  k_fps<<<B, 256, 0, stream>>>(l1xyz, distb, fidx2, l2xyz, 512, 128);
  {
    const float radii[2] = {0.4f, 0.8f};
    const int   ks[2]    = {64, 128};
    const int   base[2]  = {17, 20};
    int cofs = 0;
    for (int br = 0; br < 2; ++br){
      int K = ks[br]; float r2 = radii[br] * radii[br];
      k_ballquery<<<(B * 128 + 127) / 128, 128, 0, stream>>>(l1xyz, l2xyz, gidx, B, 512, 128, K, r2);
      int C = cfg[base[br] + 2].Nout, Win = ceil32(C);
      for (int b = 0; b < B; ++b){
        int rows = 128 * K;
        k_build_group<<<(rows + 255) / 256, 256, 0, stream>>>(l1xyz, l2xyz, gidx, l1pts,
                                                              INh, b, 512, 128, K, 320, 352);
        gemm(INh,  base[br] + 0, rows, ACTA, true, true);
        gemm(ACTA, base[br] + 1, rows, ACTB, true, true);
        gemm(ACTB, base[br] + 2, rows, ACTA, true, true);
        int tot = 128 * C;
        k_maxpool<<<(tot + 255) / 256, 256, 0, stream>>>(ACTA, l2pts + (size_t)b * 128 * 512,
                                                         128, K, C, Win, 512, cofs);
      }
      cofs += C;
    }
  }

  // ---- SA3: global MLP 515(->544) -> 256 -> 512 -> 1024, max over 128 pts
  k_build_sa3<<<(B * 128 + 255) / 256, 256, 0, stream>>>(l2xyz, l2pts, INh, B * 128, 544);
  gemm(INh,  23, B * 128, ACTA, true, true);
  gemm(ACTA, 24, B * 128, ACTB, true, true);
  gemm(ACTB, 25, B * 128, ACTA, true, true);
  k_maxpool<<<(B * 1024 + 255) / 256, 256, 0, stream>>>(ACTA, l3f16, B, 128, 1024, 1024, 1024, 0);
  k_h2f<<<(B * 1024 + 255) / 256, 256, 0, stream>>>(l3f16, (float*)d_out + (size_t)B * N * 50, B * 1024);

  // ---- FP3: [l2_points(512) | l3 broadcast(1024)] -> 256 -> 256
  k_build_fp3<<<(B * 128 + 255) / 256, 256, 0, stream>>>(l2pts, l3f16, INh, B, 128, 1536);
  gemm(INh,  6, B * 128, ACTA, true, true);
  gemm(ACTA, 7, B * 128, l2p, true, true);

  // ---- FP2: l1_points concat interp(l2p, 3NN l2->l1) -> 576 -> 256 -> 128
  k_interp_build<<<(B * 512 + 127) / 128, 128, 0, stream>>>(l1xyz, l2xyz, l1pts, 320,
                                                            l2p, 256, INh, B, 512, 128, 576);
  gemm(INh,  4, B * 512, ACTA, true, true);
  gemm(ACTA, 5, B * 512, l1p, true, true);

  // ---- FP1: p1=[cls(16)|xyz|xyz] concat interp(l1p) -> 150(->160) -> 128 -> 128
  k_build_p1<<<(B * N + 255) / 256, 256, 0, stream>>>(cls, l0xyz, p1b, B, N);
  k_interp_build<<<(B * N + 127) / 128, 128, 0, stream>>>(l0xyz, l1xyz, p1b, 22,
                                                          l1p, 128, INh, B, N, 512, 160);
  gemm(INh,  2, B * N, ACTA, true, true);
  gemm(ACTA, 3, B * N, ACTB, true, true);

  // ---- conv1 (relu) / conv2 (f32 logits) / log_softmax
  gemm(ACTB, 0, B * N, ACTA, true, true);
  gemm(ACTA, 1, B * N, logit, false, false);
  k_logsoftmax<<<(B * N + 127) / 128, 128, 0, stream>>>(logit, (float*)d_out, B * N, 50, 64);
}
